// TrajectoryVAE_29137058136630
// MI455X (gfx1250) — compile-verified, matrix operand order checked by reading, not run on hardware
//
#include <hip/hip_runtime.h>
#include <hip/hip_bf16.h>

// ---------------------------------------------------------------------------
// TrajectoryVAE forward for MI455X (gfx1250), wave32 + WMMA bf16 +
// async global->LDS copy (ASYNCcnt) with double-buffered single-barrier
// pipelines in all WMMA GEMMs.
// ---------------------------------------------------------------------------

typedef __bf16 bf16_t;
typedef __bf16 v8bf  __attribute__((ext_vector_type(8)));
typedef __bf16 v16bf __attribute__((ext_vector_type(16)));
typedef float  v8f   __attribute__((ext_vector_type(8)));

constexpr int Bc = 8192;   // batch
constexpr int Tc = 32;     // timesteps
constexpr int Dc = 12;     // event dim
constexpr int Zc = 128;    // latent
constexpr int HEc = 512;   // encoder hidden
constexpr int HDc = 1024;  // decoder hidden

__device__ __forceinline__ float sigmoidf_(float x) {
  return 1.0f / (1.0f + __expf(-x));
}

// Lower 32 bits of a generic LDS pointer = workgroup-relative LDS offset
// (ISA 10.2 aperture mapping).
__device__ __forceinline__ unsigned lds_u32(const void* p) {
  return (unsigned)(uintptr_t)p;
}

// Per-lane 16B async copy global -> LDS (tracked with ASYNCcnt).
__device__ __forceinline__ void async_b128(unsigned lds, const bf16_t* g) {
  asm volatile("global_load_async_to_lds_b128 %0, %1, off"
               :
               : "v"(lds), "v"(g)
               : "memory");
}
__device__ __forceinline__ void wait_async0() {
  asm volatile("s_wait_asynccnt 0x0" ::: "memory");
}

// Build a 16-element bf16 WMMA fragment from a row-major LDS row:
// per ISA 7.12.2, lane holds K = {kb..kb+7, kb+16..kb+23} of its row.
__device__ __forceinline__ v16bf frag16(const bf16_t* p) {
  v8bf lo = *(const v8bf*)p;
  v8bf hi = *(const v8bf*)(p + 16);
  return __builtin_shufflevector(lo, hi, 0, 1, 2, 3, 4, 5, 6, 7,
                                 8, 9, 10, 11, 12, 13, 14, 15);
}

// ---------------------------------------------------------------------------
// Fused LSTM step: gates = h_prev @ Whh^T (WMMA bf16) + x_t @ Wih^T + b,
// then c/h update entirely in registers (i,f,g,o co-located per lane).
// Block: 256 threads = 8 waves. Tile: 128 batch rows x (4 gates x 32 cols).
// Wave w owns rows [w*16, w*16+16) and all 8 N-tiles of the 128-wide tile.
// ---------------------------------------------------------------------------
template <int H>
__global__ __launch_bounds__(256) void lstm_step_kernel(
    const bf16_t* __restrict__ h_prev,  // (B,H) bf16
    float* __restrict__ c_state,        // (B,H) f32 in/out
    bf16_t* __restrict__ h_next,        // (B,H) bf16
    const bf16_t* __restrict__ Whh,     // (4H,H) bf16 row-major
    const float* __restrict__ Wih,      // (4H,D) f32
    const float* __restrict__ bias,     // (4H) f32
    const float* __restrict__ x,        // (B,T,D) f32
    const float* __restrict__ start_tok,// (D) f32
    int xt,                             // timestep into x; <0 => start token
    const int* __restrict__ lengths,    // encoder only (else null)
    float* __restrict__ h_n,            // encoder capture (B,H) f32 (else null)
    int t) {
  constexpr int LDA = 40;  // 80B row stride: 16B aligned, spreads banks
  __shared__ __align__(16) bf16_t aT[2][128 * LDA];
  __shared__ __align__(16) bf16_t bT[2][128 * LDA];
  __shared__ float xT[128 * Dc];
  __shared__ float wT[128 * Dc];
  __shared__ float bsT[128];

  const int tid = threadIdx.x;
  const int lane = tid & 31;
  const int wv = tid >> 5;
  const int m0 = blockIdx.x * 128;
  const int n0 = blockIdx.y * 32;

  // Stage x_t rows, Wih slice (gate-major remap), bias slice (epilogue data).
  for (int i = tid; i < 128 * Dc; i += 256) {
    int r = i / Dc, d = i - r * Dc;
    xT[i] = (xt < 0) ? start_tok[d]
                     : x[((size_t)(m0 + r) * Tc + xt) * Dc + d];
    int q = r >> 5, c = r & 31;
    wT[i] = Wih[(size_t)(q * H + n0 + c) * Dc + d];
  }
  if (tid < 128) {
    int q = tid >> 5, c = tid & 31;
    bsT[tid] = bias[q * H + n0 + c];
  }

  // Issue one K-chunk (128x32 of A and of gate-remapped B) as 4 async ops
  // per lane: chunks c = tid and tid+256; chunk c -> row c>>2, 16B seg c&3.
  const int sr = tid >> 2;   // row for chunk tid (0..63)
  const int ss = tid & 3;    // 16B segment
  auto stage_async = [&](int buf, int k0) {
    {
      int r = sr;
      async_b128(lds_u32(&aT[buf][r * LDA + ss * 8]),
                 &h_prev[(size_t)(m0 + r) * H + k0 + ss * 8]);
      int q = r >> 5, col = r & 31;
      async_b128(lds_u32(&bT[buf][r * LDA + ss * 8]),
                 &Whh[(size_t)(q * H + n0 + col) * H + k0 + ss * 8]);
    }
    {
      int r = sr + 64;
      async_b128(lds_u32(&aT[buf][r * LDA + ss * 8]),
                 &h_prev[(size_t)(m0 + r) * H + k0 + ss * 8]);
      int q = r >> 5, col = r & 31;
      async_b128(lds_u32(&bT[buf][r * LDA + ss * 8]),
                 &Whh[(size_t)(q * H + n0 + col) * H + k0 + ss * 8]);
    }
  };

  v8f acc[8];
#pragma unroll
  for (int nt = 0; nt < 8; nt++)
#pragma unroll
    for (int e = 0; e < 8; e++) acc[nt][e] = 0.0f;

  const int fr = lane & 15;
  const int kb = (lane >> 4) * 8;

  // Pipeline: wait own async -> barrier (publish) -> prefetch next -> compute.
  stage_async(0, 0);
  int cur = 0;
  for (int k0 = 0; k0 < H; k0 += 32) {
    wait_async0();
    __syncthreads();
    if (k0 + 32 < H) stage_async(cur ^ 1, k0 + 32);
    v16bf a = frag16(&aT[cur][(wv * 16 + fr) * LDA + kb]);
#pragma unroll
    for (int nt = 0; nt < 8; nt++) {
      v16bf b = frag16(&bT[cur][(nt * 16 + fr) * LDA + kb]);
      acc[nt] = __builtin_amdgcn_wmma_f32_16x16x32_bf16(
          false, a, false, b, (short)0, acc[nt], false, false);
    }
    cur ^= 1;
  }

  // Epilogue: acc[2q+j][v] holds gate q, col n0+j*16+(lane&15),
  // batch row m0 + wv*16 + v + 8*(lane>>4).  i/f/g/o share lane+element.
  const int rsub = (lane >> 4) * 8;
  const int cl = lane & 15;
#pragma unroll
  for (int j = 0; j < 2; j++) {
#pragma unroll
    for (int v = 0; v < 8; v++) {
      int r = wv * 16 + v + rsub;
      int lc = j * 16 + cl;
      int bi = m0 + r;
      int col = n0 + lc;
      float g4[4];
#pragma unroll
      for (int q = 0; q < 4; q++) {
        float s = bsT[q * 32 + lc] + acc[2 * q + j][v];
        const float* wr = &wT[(q * 32 + lc) * Dc];
        const float* xr = &xT[r * Dc];
#pragma unroll
        for (int d = 0; d < Dc; d++) s += xr[d] * wr[d];
        g4[q] = s;
      }
      float iv = sigmoidf_(g4[0]);
      float fv = sigmoidf_(g4[1]);
      float gv = tanhf(g4[2]);
      float ov = sigmoidf_(g4[3]);
      size_t idx = (size_t)bi * H + col;
      float cn = fv * c_state[idx] + iv * gv;
      c_state[idx] = cn;
      float hn = ov * tanhf(cn);
      h_next[idx] = (bf16_t)hn;
      if (h_n != nullptr) {
        int L = lengths[bi];
        L = L < 1 ? 1 : (L > Tc ? Tc : L);
        if (L - 1 == t) h_n[idx] = hn;
      }
    }
  }
}

// ---------------------------------------------------------------------------
// Output projection: x_recon = hs_d @ W_out^T + b_out, N padded 12->16, WMMA.
// Whole padded W_out (16x1024 bf16 = 32KB) is LDS-persistent (loaded once,
// row stride 1032 elems => 16B-aligned chunks, rows 4 banks apart).
// A tile is async double-buffered, single barrier per K-chunk.
// ---------------------------------------------------------------------------
__global__ __launch_bounds__(256) void out_proj_kernel(
    const bf16_t* __restrict__ hs,   // (T*B, HD) bf16
    const bf16_t* __restrict__ WoP,  // (16, HD) bf16 (rows 12..15 zero)
    const float* __restrict__ b_out, // (D)
    float* __restrict__ xrec) {      // (B,T,D)
  constexpr int LDA = 40;
  constexpr int LDW = HDc + 8;  // 1032
  __shared__ __align__(16) bf16_t aT[2][128 * LDA];
  __shared__ __align__(16) bf16_t wL[16 * LDW];
  const int tid = threadIdx.x, lane = tid & 31, wv = tid >> 5;
  const size_t r0 = (size_t)blockIdx.x * 128;

  // Persistent B: load all 16x1024 once (8 chunks of 8 bf16 per thread).
  for (int i = tid * 8; i < 16 * HDc; i += 256 * 8) {
    int r = i >> 10, k = i & (HDc - 1);
    *(v8bf*)&wL[r * LDW + k] = *(const v8bf*)&WoP[(size_t)r * HDc + k];
  }

  const int sr = tid >> 2;  // A-chunk row (0..63)
  const int ss = tid & 3;   // 16B segment
  auto stage_a = [&](int buf, int k0) {
    async_b128(lds_u32(&aT[buf][sr * LDA + ss * 8]),
               &hs[(r0 + sr) * HDc + k0 + ss * 8]);
    async_b128(lds_u32(&aT[buf][(sr + 64) * LDA + ss * 8]),
               &hs[(r0 + sr + 64) * HDc + k0 + ss * 8]);
  };

  v8f acc;
#pragma unroll
  for (int e = 0; e < 8; e++) acc[e] = 0.0f;
  const int fr = lane & 15;
  const int kb = (lane >> 4) * 8;

  stage_a(0, 0);
  int cur = 0;
  for (int k0 = 0; k0 < HDc; k0 += 32) {
    wait_async0();
    __syncthreads();
    if (k0 + 32 < HDc) stage_a(cur ^ 1, k0 + 32);
    v16bf a = frag16(&aT[cur][(wv * 16 + fr) * LDA + kb]);
    v16bf b = frag16(&wL[fr * LDW + k0 + kb]);
    acc = __builtin_amdgcn_wmma_f32_16x16x32_bf16(false, a, false, b, (short)0,
                                                  acc, false, false);
    cur ^= 1;
  }
  const int cl = lane & 15;
  if (cl < Dc) {
    const int rsub = (lane >> 4) * 8;
    float bo = b_out[cl];
#pragma unroll
    for (int v = 0; v < 8; v++) {
      size_t R = r0 + wv * 16 + v + rsub;  // row in (T,B) order
      int tt = (int)(R >> 13);             // / 8192
      int bb = (int)(R & 8191);
      xrec[((size_t)bb * Tc + tt) * Dc + cl] = acc[v] + bo;
    }
  }
}

// ---------------------------------------------------------------------------
// mu/logvar/z (K=512, N=128 per side): L2-resident weights, VALU.
// ---------------------------------------------------------------------------
__global__ __launch_bounds__(128) void proj_z_kernel(
    const float* __restrict__ hN, const float* __restrict__ Wmu,
    const float* __restrict__ bmu, const float* __restrict__ Wlv,
    const float* __restrict__ blv, const float* __restrict__ eps,
    float* __restrict__ out_mu, float* __restrict__ out_lv,
    float* __restrict__ out_z, bf16_t* __restrict__ zbf) {
  __shared__ float hrow[HEc];
  const int b = blockIdx.x, j = threadIdx.x;
  for (int i = j; i < HEc; i += 128) hrow[i] = hN[(size_t)b * HEc + i];
  __syncthreads();
  float mu = bmu[j], lv = blv[j];
  const float* wm = &Wmu[(size_t)j * HEc];
  const float* wl = &Wlv[(size_t)j * HEc];
  for (int k = 0; k < HEc; k++) {
    float h = hrow[k];
    mu += h * wm[k];
    lv += h * wl[k];
  }
  float zz = mu + eps[(size_t)b * Zc + j] * __expf(0.5f * lv);
  size_t o = (size_t)b * Zc + j;
  out_mu[o] = mu;
  out_lv[o] = lv;
  out_z[o] = zz;
  zbf[o] = (bf16_t)zz;
}

// h0/c0 = tanh(z @ W^T + b), K=128: VALU, weights L2-resident.
__global__ __launch_bounds__(256) void init_dec_kernel(
    const float* __restrict__ z, const float* __restrict__ Wh0,
    const float* __restrict__ bh0, const float* __restrict__ Wc0,
    const float* __restrict__ bc0, bf16_t* __restrict__ h0,
    float* __restrict__ c0) {
  __shared__ float zr[Zc];
  const int b = blockIdx.x;
  for (int i = threadIdx.x; i < Zc; i += 256) zr[i] = z[(size_t)b * Zc + i];
  __syncthreads();
  for (int j = threadIdx.x; j < HDc; j += 256) {
    float h = bh0[j], c = bc0[j];
    const float* wh = &Wh0[(size_t)j * Zc];
    const float* wc = &Wc0[(size_t)j * Zc];
    for (int k = 0; k < Zc; k++) {
      float zv = zr[k];
      h += zv * wh[k];
      c += zv * wc[k];
    }
    h0[(size_t)b * HDc + j] = (bf16_t)tanhf(h);
    c0[(size_t)b * HDc + j] = tanhf(c);
  }
}

// ------------------------------- utilities ---------------------------------
__global__ void k_cast_bf16(bf16_t* __restrict__ dst,
                            const float* __restrict__ src, int n) {
  for (int i = blockIdx.x * blockDim.x + threadIdx.x; i < n;
       i += gridDim.x * blockDim.x)
    dst[i] = (bf16_t)src[i];
}
__global__ void k_zero_bf16(bf16_t* __restrict__ dst, int n) {
  for (int i = blockIdx.x * blockDim.x + threadIdx.x; i < n;
       i += gridDim.x * blockDim.x)
    dst[i] = (bf16_t)0.0f;
}
__global__ void k_zero_f32(float* __restrict__ dst, int n) {
  for (int i = blockIdx.x * blockDim.x + threadIdx.x; i < n;
       i += gridDim.x * blockDim.x)
    dst[i] = 0.0f;
}
__global__ void k_pack_wout(bf16_t* __restrict__ dst,
                            const float* __restrict__ W, int K) {
  int n = 16 * K;
  for (int i = blockIdx.x * blockDim.x + threadIdx.x; i < n;
       i += gridDim.x * blockDim.x) {
    int r = i / K, k = i - r * K;
    dst[i] = (r < Dc) ? (bf16_t)W[r * K + k] : (bf16_t)0.0f;
  }
}

// ---------------------------------------------------------------------------
extern "C" void kernel_launch(void* const* d_in, const int* in_sizes, int n_in,
                              void* d_out, int out_size, void* d_ws,
                              size_t ws_size, hipStream_t stream) {
  (void)in_sizes; (void)n_in; (void)out_size; (void)ws_size;

  const float* x       = (const float*)d_in[0];
  const int*   lengths = (const int*)d_in[1];
  const float* eps     = (const float*)d_in[2];
  const float* W_ih_e  = (const float*)d_in[3];
  const float* W_hh_e  = (const float*)d_in[4];
  const float* b_e     = (const float*)d_in[5];
  const float* W_mu    = (const float*)d_in[6];
  const float* b_mu    = (const float*)d_in[7];
  const float* W_lv    = (const float*)d_in[8];
  const float* b_lv    = (const float*)d_in[9];
  const float* W_h0    = (const float*)d_in[10];
  const float* b_h0    = (const float*)d_in[11];
  const float* W_c0    = (const float*)d_in[12];
  const float* b_c0    = (const float*)d_in[13];
  const float* start_t = (const float*)d_in[14];
  const float* W_ih_d  = (const float*)d_in[15];
  const float* W_hh_d  = (const float*)d_in[16];
  const float* b_d     = (const float*)d_in[17];
  const float* W_out   = (const float*)d_in[18];
  const float* b_out   = (const float*)d_in[19];

  float* out_xrec = (float*)d_out;                      // (B,T,D)
  float* out_mu   = out_xrec + (size_t)Bc * Tc * Dc;    // (B,Z)
  float* out_lv   = out_mu + (size_t)Bc * Zc;           // (B,Z)
  float* out_z    = out_lv + (size_t)Bc * Zc;           // (B,Z)

  // Workspace carve-out (256B aligned slabs).
  char* wsp = (char*)d_ws;
  size_t off = 0;
  auto take = [&](size_t bytes) -> char* {
    char* p = wsp + off;
    off += (bytes + 255) & ~(size_t)255;
    return p;
  };
  bf16_t* whhE  = (bf16_t*)take((size_t)4 * HEc * HEc * sizeof(bf16_t));
  bf16_t* whhD  = (bf16_t*)take((size_t)4 * HDc * HDc * sizeof(bf16_t));
  bf16_t* woutP = (bf16_t*)take((size_t)16 * HDc * sizeof(bf16_t));
  bf16_t* hPing = (bf16_t*)take((size_t)Bc * HEc * sizeof(bf16_t));
  bf16_t* hPong = (bf16_t*)take((size_t)Bc * HEc * sizeof(bf16_t));
  float*  cE    = (float*)take((size_t)Bc * HEc * sizeof(float));
  float*  hN    = (float*)take((size_t)Bc * HEc * sizeof(float));
  bf16_t* zbf   = (bf16_t*)take((size_t)Bc * Zc * sizeof(bf16_t));
  bf16_t* h0d   = (bf16_t*)take((size_t)Bc * HDc * sizeof(bf16_t));
  float*  cD    = (float*)take((size_t)Bc * HDc * sizeof(float));
  bf16_t* hsD   = (bf16_t*)take((size_t)Tc * Bc * HDc * sizeof(bf16_t));

  auto blocks = [](int n) { return dim3((unsigned)((n + 255) / 256)); };

  // Prep: weight casts + state init (every call; deterministic).
  int n;
  n = 4 * HEc * HEc; k_cast_bf16<<<blocks(n), 256, 0, stream>>>(whhE, W_hh_e, n);
  n = 4 * HDc * HDc; k_cast_bf16<<<blocks(n), 256, 0, stream>>>(whhD, W_hh_d, n);
  k_pack_wout<<<blocks(16 * HDc), 256, 0, stream>>>(woutP, W_out, HDc);
  n = Bc * HEc;
  k_zero_bf16<<<blocks(n), 256, 0, stream>>>(hPing, n);
  k_zero_f32<<<blocks(n), 256, 0, stream>>>(cE, n);

  // Encoder scan (ping-pong h to avoid intra-step RAW across workgroups).
  bf16_t* hp = hPing;
  bf16_t* hq = hPong;
  for (int t = 0; t < Tc; t++) {
    lstm_step_kernel<HEc><<<dim3(Bc / 128, HEc / 32), 256, 0, stream>>>(
        hp, cE, hq, whhE, W_ih_e, b_e, x, start_t, t, lengths, hN, t);
    bf16_t* tmp = hp; hp = hq; hq = tmp;
  }

  // Latent head + decoder init.
  proj_z_kernel<<<Bc, 128, 0, stream>>>(hN, W_mu, b_mu, W_lv, b_lv, eps,
                                        out_mu, out_lv, out_z, zbf);
  init_dec_kernel<<<Bc, 256, 0, stream>>>(out_z, W_h0, b_h0, W_c0, b_c0, h0d,
                                          cD);

  // Decoder scan: hidden states stream into hsD (natural double-buffering).
  for (int t = 0; t < Tc; t++) {
    const bf16_t* hprev = (t == 0) ? h0d : hsD + (size_t)(t - 1) * Bc * HDc;
    lstm_step_kernel<HDc><<<dim3(Bc / 128, HDc / 32), 256, 0, stream>>>(
        hprev, cD, hsD + (size_t)t * Bc * HDc, whhD, W_ih_d, b_d, x, start_t,
        t - 1, nullptr, nullptr, t);
  }

  // Output projection (WMMA, N padded to 16).
  out_proj_kernel<<<(Tc * Bc) / 128, 256, 0, stream>>>(hsD, woutP, b_out,
                                                       out_xrec);
}
